// SparseMaskConv_33913061769460
// MI455X (gfx1250) — matrix operand order repaired and checked by reading, not compiled, not run on hardware
//
#include <hip/hip_runtime.h>
#include <math.h>

// ---------------------------------------------------------------------------
// MI455X / gfx1250 implementation of the LSH sparse-attention reference.
// All heavy math runs on v_wmma_f32_16x16x32_bf16 (fp32 accumulate).
// Working set ~50MB -> L2 resident; compute-bound -> bf16 WMMA path.
// ---------------------------------------------------------------------------

#define L     16384
#define C     64
#define CR    16
#define NH    4
#define HB    32
#define CHUNK 512
#define NS    65536      // NH * L sorted entries
#define RES_SCALE 0.1f

typedef __attribute__((ext_vector_type(16))) __bf16 v16bf;
typedef __attribute__((ext_vector_type(8)))  float  v8f;

union Frag { v16bf v; uint4 q[2]; };

__device__ inline v8f v8zero() { v8f z; for (int i = 0; i < 8; ++i) z[i] = 0.f; return z; }

// float -> bf16 (round to nearest even), stored as raw ushort
__device__ inline unsigned short f2bf(float f) {
  unsigned u = __float_as_uint(f);
  u += 0x7fffu + ((u >> 16) & 1u);
  return (unsigned short)(u >> 16);
}

// A-fragment (16x32 bf16, M x K), src row-major with rowStride ushorts.
// ISA layout: lane<16: m=lane, k in [kb..kb+7] & [16+kb..16+kb+7], kb=0;
//             lane>=16: m=lane-16, kb=8.
__device__ inline v16bf loadA(const unsigned short* tile, int rowStride) {
  int l = threadIdx.x & 31;
  int m = l & 15;
  int kb = (l < 16) ? 0 : 8;
  Frag f;
  const unsigned short* p = tile + m * rowStride;
  f.q[0] = *(const uint4*)(p + kb);
  f.q[1] = *(const uint4*)(p + 16 + kb);
  return f.v;
}

// B-fragment (32x16 bf16, K x N), src stored n-major: element(k,n) at
// base[n*nStride + k].  lane<16: n=lane, k=0..7,16..23; lane>=16: n=lane-16, k=8..15,24..31.
__device__ inline v16bf loadB(const unsigned short* base, int nStride) {
  int l = threadIdx.x & 31;
  int n = l & 15;
  int ko = (l < 16) ? 0 : 8;
  Frag f;
  const unsigned short* p = base + n * nStride + ko;
  f.q[0] = *(const uint4*)(p);
  f.q[1] = *(const uint4*)(p + 16);
  return f.v;
}

__device__ inline v8f wmma_bf16(v16bf a, v16bf b, v8f c) {
  return __builtin_amdgcn_wmma_f32_16x16x32_bf16(false, a, false, b, (short)0, c, false, false);
}

// ------------------------------ K0: transpose x [C][L] -> xf [L][C] ---------
__global__ void __launch_bounds__(256) k_transpose(const float* __restrict__ x,
                                                   float* __restrict__ xf) {
  int t = blockIdx.x * 256 + threadIdx.x;
  for (int c = 0; c < C; ++c) xf[(size_t)t * C + c] = x[(size_t)c * L + t];
}

// ------------------------------ K1: projections q = xf@Wm^T+bm, y = xf@Wa^T+ba
// q kept fp32 (feeds exact hash argmax), y stored bf16 token-major.
__global__ void __launch_bounds__(256) k_proj(const float* __restrict__ xf,
    const float* __restrict__ wm, const float* __restrict__ bm,
    const float* __restrict__ wa, const float* __restrict__ ba,
    float* __restrict__ q32, unsigned short* __restrict__ ybf) {
  __shared__ unsigned short sX[256 * 64];   // 32KB token-major bf16 tile
  __shared__ unsigned short sWm[16 * 64];   // 2KB
  __shared__ unsigned short sWa[64 * 64];   // 8KB
  int tid = threadIdx.x;
  int t0 = blockIdx.x * 256;
  for (int i = tid; i < 256 * 64; i += 256) sX[i] = f2bf(xf[(size_t)t0 * 64 + i]);
  for (int i = tid; i < 16 * 64;  i += 256) sWm[i] = f2bf(wm[i]);
  for (int i = tid; i < 64 * 64;  i += 256) sWa[i] = f2bf(wa[i]);
  __syncthreads();
  int wv = tid >> 5, l = tid & 31;
  int half8 = (l < 16) ? 0 : 8;
  for (int job = wv; job < 80; job += 8) {       // 16 M-tiles x (1 q + 4 y jobs)
    int mt = job / 5, jo = job % 5;
    const unsigned short* Abase = sX + (mt * 16) * 64;
    const unsigned short* W = (jo == 0) ? sWm : (sWa + (jo - 1) * 16 * 64);
    v8f acc = v8zero();
    for (int ks = 0; ks < 2; ++ks) {             // K=64 -> 2 bf16 WMMAs
      v16bf a = loadA(Abase + ks * 32, 64);
      v16bf b = loadB(W + ks * 32, 64);
      acc = wmma_bf16(a, b, acc);
    }
    int rbase = t0 + mt * 16 + half8;
    if (jo == 0) {
      float bias = bm[l & 15];
      for (int v = 0; v < 8; ++v) q32[(size_t)(rbase + v) * CR + (l & 15)] = acc[v] + bias;
    } else {
      int col = (jo - 1) * 16 + (l & 15);
      float bias = ba[col];
      for (int v = 0; v < 8; ++v) ybf[(size_t)(rbase + v) * C + col] = f2bf(acc[v] + bias);
    }
  }
}

// ------------------------------ K2: LSH codes -------------------------------
__global__ void __launch_bounds__(256) k_hash(const float* __restrict__ q32,
    const float* __restrict__ rot,    // [CR][NH][HB/2] for this layer
    int* __restrict__ codes) {
  __shared__ float sR[CR * NH * 16];  // 4KB
  int tid = threadIdx.x;
  for (int i = tid; i < CR * NH * 16; i += 256) sR[i] = rot[i];
  __syncthreads();
  int t = blockIdx.x * 256 + tid;
  float qv[16];
  for (int f = 0; f < 16; ++f) qv[f] = q32[(size_t)t * CR + f];
  for (int h = 0; h < NH; ++h) {
    float rv[16];
    for (int i = 0; i < 16; ++i) {
      float a = 0.f;
      for (int f = 0; f < 16; ++f) a += qv[f] * sR[f * 64 + h * 16 + i];
      rv[i] = a;
    }
    float best = -INFINITY; int bi = 0;
    for (int i = 0; i < 32; ++i) {               // argmax over [rv, -rv]
      float v = (i < 16) ? rv[i] : -rv[i - 16];
      if (v > best) { best = v; bi = i; }
    }
    codes[h * L + t] = bi + h * HB;
  }
}

// ------------------------------ K3: stable counting sort (128 bins) ---------
__global__ void __launch_bounds__(512) k_hist(const int* __restrict__ codes,
                                              int* __restrict__ bh) {
  __shared__ int h[128];
  int tid = threadIdx.x;
  if (tid < 128) h[tid] = 0;
  __syncthreads();
  atomicAdd(&h[codes[blockIdx.x * 512 + tid]], 1);
  __syncthreads();
  if (tid < 128) bh[blockIdx.x * 128 + tid] = h[tid];
}

__global__ void __launch_bounds__(128) k_offsets(const int* __restrict__ bh,
                                                 int* __restrict__ bstart) {
  __shared__ int tot[128];
  __shared__ int binoff[128];
  int b = threadIdx.x;
  int run = 0;
  for (int blk = 0; blk < 128; ++blk) { bstart[blk * 128 + b] = run; run += bh[blk * 128 + b]; }
  tot[b] = run;
  __syncthreads();
  if (b == 0) { int s = 0; for (int i = 0; i < 128; ++i) { binoff[i] = s; s += tot[i]; } }
  __syncthreads();
  int off = binoff[b];
  for (int blk = 0; blk < 128; ++blk) bstart[blk * 128 + b] += off;
}

__global__ void __launch_bounds__(512) k_scatter(const int* __restrict__ codes,
    const int* __restrict__ bstart, int* __restrict__ idx, int* __restrict__ undo) {
  __shared__ int sc[512];
  int tid = threadIdx.x;
  int e = blockIdx.x * 512 + tid;
  int c = codes[e];
  sc[tid] = c;
  __syncthreads();
  int rank = 0;
  for (int j = 0; j < tid; ++j) rank += (sc[j] == c);   // stable intra-block rank
  int p = bstart[blockIdx.x * 128 + c] + rank;
  idx[p] = e;
  undo[e] = p;
}

// ------------------------------ K4: gather sorted Q/K/V ---------------------
// qA: unnormalized Q, [NS][32] bf16 (K padded 16->32 with zeros)
// qK: normalized K,   [NS][32] bf16
// ykT: V transposed per (hash,chunk): [hk][64][512] bf16
__global__ void __launch_bounds__(256) k_gather(const int* __restrict__ idx,
    const float* __restrict__ q32, const unsigned short* __restrict__ ybf,
    unsigned short* __restrict__ qA, unsigned short* __restrict__ qK,
    unsigned short* __restrict__ ykT) {
  int s = blockIdx.x * 256 + threadIdx.x;
  int t = idx[s] & (L - 1);                     // idx % L
  float qv[16]; float nrm = 0.f;
  for (int f = 0; f < 16; ++f) { qv[f] = q32[(size_t)t * CR + f]; nrm += qv[f] * qv[f]; }
  float inv = 1.0f / fmaxf(sqrtf(nrm), 5e-5f);
  for (int f = 0; f < 16; ++f) {
    qA[(size_t)s * 32 + f] = f2bf(qv[f]);
    qK[(size_t)s * 32 + f] = f2bf(qv[f] * inv);
  }
  for (int f = 16; f < 32; ++f) { qA[(size_t)s * 32 + f] = 0; qK[(size_t)s * 32 + f] = 0; }
  int hk = s >> 9, i = s & 511;
  for (int c = 0; c < C; ++c)
    ykT[((size_t)hk * 64 + c) * 512 + i] = ybf[(size_t)t * C + c];
}

// ------------------------------ K5: chunk attention (flash, 2-pass) ---------
// One workgroup (8 waves) per (hash, chunk).  LDS: K 96KB + V 192KB + P 8KB.
__global__ void __launch_bounds__(256) k_attn(const unsigned short* __restrict__ qA,
    const unsigned short* __restrict__ qK, const unsigned short* __restrict__ ykT,
    float* __restrict__ ret, float* __restrict__ lseg) {
  extern __shared__ unsigned short smem[];
  unsigned short* Kb  = smem;                    // [1536][32]
  unsigned short* Yk  = smem + 1536 * 32;        // [64][1536]  (n-major for B frags)
  unsigned short* Pst = Yk + 64 * 1536;          // 8 waves * [16][32]
  int tid = threadIdx.x;
  int hk = blockIdx.x;
  int h = hk >> 5, k = hk & 31;
  int cs0 = k, cs1 = (k + 31) & 31, cs2 = (k + 1) & 31;  // self, prev, next

  // K tile: 3 chunks x 512 rows x 32 bf16 (uint4 = 8 bf16, 4 per row)
  for (int j = tid; j < 3 * 2048; j += 256) {
    int sl = j >> 11, r = j & 2047;
    int ck = (sl == 0) ? cs0 : (sl == 1) ? cs1 : cs2;
    ((uint4*)Kb)[sl * 2048 + r] =
        ((const uint4*)qK)[((size_t)(h * 32 + ck) * 512) * 4 + r];
  }
  // V tile: Yk[c][sl*512 + i]  <- ykT[(h*32+ck)*64 + c][i]
  for (int j = tid; j < 3 * 64 * 64; j += 256) {
    int sl = j / 4096, rem = j - sl * 4096;
    int c = rem >> 6, u = rem & 63;
    int ck = (sl == 0) ? cs0 : (sl == 1) ? cs1 : cs2;
    ((uint4*)Yk)[c * 192 + sl * 64 + u] =
        ((const uint4*)ykT)[((size_t)(h * 32 + ck) * 64 + c) * 64 + u];
  }
  __syncthreads();

  int wv = tid >> 5, l = tid & 31;
  unsigned short* stage = Pst + wv * 512;        // per-wave P staging [16][32]
  int half8 = (l < 16) ? 0 : 8;

  for (int mt = wv * 4; mt < wv * 4 + 4; ++mt) {   // 4 M-tiles per wave
    v16bf aq = loadA(qA + (size_t)(hk * 512 + mt * 16) * 32, 32);

    // ---- pass 1: online logsumexp over 96 j-tiles -------------------------
    float mrun[8], srun[8];
    for (int v = 0; v < 8; ++v) { mrun[v] = -INFINITY; srun[v] = 0.f; }
    for (int jt = 0; jt < 96; ++jt) {
      v16bf bk = loadB(Kb + jt * 16 * 32, 32);
      v8f S = wmma_bf16(aq, bk, v8zero());
      for (int v = 0; v < 8; ++v) {
        float x = S[v];
        float tmax = x;
        for (int msk = 1; msk < 16; msk <<= 1)
          tmax = fmaxf(tmax, __shfl_xor(tmax, msk, 32));
        float nm = fmaxf(mrun[v], tmax);
        float ts = __expf(x - nm);
        for (int msk = 1; msk < 16; msk <<= 1)
          ts += __shfl_xor(ts, msk, 32);
        srun[v] = srun[v] * __expf(mrun[v] - nm) + ts;
        mrun[v] = nm;
      }
    }
    float lse[8];
    for (int v = 0; v < 8; ++v) lse[v] = mrun[v] + __logf(srun[v]);
    if ((l & 15) == 0)
      for (int v = 0; v < 8; ++v)
        lseg[hk * 512 + mt * 16 + half8 + v] = lse[v];

    // ---- pass 2: O += exp(S - lse) @ V, 48 j-steps of 32 ------------------
    v8f O0 = v8zero(), O1 = v8zero(), O2 = v8zero(), O3 = v8zero();
    for (int js = 0; js < 48; ++js) {
      v16bf bk0 = loadB(Kb + (2 * js) * 16 * 32, 32);
      v16bf bk1 = loadB(Kb + (2 * js + 1) * 16 * 32, 32);
      v8f S0 = wmma_bf16(aq, bk0, v8zero());
      v8f S1 = wmma_bf16(aq, bk1, v8zero());
      int col = l & 15;
      for (int v = 0; v < 8; ++v) {
        int r = v + half8;                       // C layout -> row-major staging
        stage[r * 32 + col]      = f2bf(__expf(S0[v] - lse[v]));
        stage[r * 32 + 16 + col] = f2bf(__expf(S1[v] - lse[v]));
      }
      // same-wave LDS RAW: compiler orders via dscnt, no barrier needed
      v16bf ap = loadA(stage, 32);
      O0 = wmma_bf16(ap, loadB(Yk + 0  * 1536 + js * 32, 1536), O0);
      O1 = wmma_bf16(ap, loadB(Yk + 16 * 1536 + js * 32, 1536), O1);
      O2 = wmma_bf16(ap, loadB(Yk + 32 * 1536 + js * 32, 1536), O2);
      O3 = wmma_bf16(ap, loadB(Yk + 48 * 1536 + js * 32, 1536), O3);
    }
    for (int v = 0; v < 8; ++v) {
      size_t row = (size_t)(hk * 512 + mt * 16 + half8 + v) * 64;
      ret[row +  0 + (l & 15)] = O0[v];
      ret[row + 16 + (l & 15)] = O1[v];
      ret[row + 32 + (l & 15)] = O2[v];
      ret[row + 48 + (l & 15)] = O3[v];
    }
  }
}

// ------------------------------ K6: unsort + hash softmax + residual --------
__global__ void __launch_bounds__(256) k_combine(const int* __restrict__ undo,
    const float* __restrict__ ret, const float* __restrict__ lseg,
    const float* __restrict__ fin, float* __restrict__ fout) {
  int t = blockIdx.x * 256 + threadIdx.x;
  int p[4]; float w[4];
  float mx = -INFINITY;
  for (int h = 0; h < 4; ++h) { p[h] = undo[h * L + t]; w[h] = lseg[p[h]]; mx = fmaxf(mx, w[h]); }
  float s = 0.f;
  for (int h = 0; h < 4; ++h) { w[h] = __expf(w[h] - mx); s += w[h]; }
  float inv = 1.0f / s;
  for (int c = 0; c < C; c += 4) {
    float4 acc = {0.f, 0.f, 0.f, 0.f};
    for (int h = 0; h < 4; ++h) {
      float4 r = *(const float4*)(ret + (size_t)p[h] * 64 + c);
      float ww = w[h] * inv;
      acc.x += ww * r.x; acc.y += ww * r.y; acc.z += ww * r.z; acc.w += ww * r.w;
    }
    float4 xi = *(const float4*)(fin + (size_t)t * C + c);
    float4 o = { acc.x * RES_SCALE + xi.x, acc.y * RES_SCALE + xi.y,
                 acc.z * RES_SCALE + xi.z, acc.w * RES_SCALE + xi.w };
    *(float4*)(fout + (size_t)t * C + c) = o;
  }
}

// ------------------------------ K7: final collect GEMM ----------------------
// out[o][t] = sum_c cat[t][c] * w_collect[o][c] + b_collect[o], cat = [f1|f2]
__global__ void __launch_bounds__(256) k_collect(const float* __restrict__ f1,
    const float* __restrict__ f2, const float* __restrict__ wc,
    const float* __restrict__ bc, float* __restrict__ out) {
  __shared__ unsigned short sCat[128 * 128];  // 32KB token-major
  __shared__ unsigned short sW[64 * 128];     // 16KB
  int tid = threadIdx.x;
  int t0 = blockIdx.x * 128;
  for (int i = tid; i < 128 * 64; i += 256) {
    int tk = i >> 6, c = i & 63;
    sCat[tk * 128 + c]      = f2bf(f1[(size_t)(t0 + tk) * 64 + c]);
    sCat[tk * 128 + 64 + c] = f2bf(f2[(size_t)(t0 + tk) * 64 + c]);
  }
  for (int i = tid; i < 64 * 128; i += 256) sW[i] = f2bf(wc[i]);
  __syncthreads();
  int wv = tid >> 5, l = tid & 31;
  int half8 = (l < 16) ? 0 : 8;
  for (int job = wv; job < 32; job += 8) {     // 8 M-tiles x 4 N-tiles
    int mt = job >> 2, nt = job & 3;
    v8f acc = v8zero();
    for (int ks = 0; ks < 4; ++ks) {           // K=128 -> 4 bf16 WMMAs
      v16bf a = loadA(sCat + (mt * 16) * 128 + ks * 32, 128);
      v16bf b = loadB(sW + (nt * 16) * 128 + ks * 32, 128);
      acc = wmma_bf16(a, b, acc);
    }
    int o = nt * 16 + (l & 15);
    float bias = bc[o];
    int tb = t0 + mt * 16 + half8;
    for (int v = 0; v < 8; ++v)
      out[(size_t)o * L + tb + v] = acc[v] + bias;
  }
}

// ---------------------------------------------------------------------------
extern "C" void kernel_launch(void* const* d_in, const int* in_sizes, int n_in,
                              void* d_out, int out_size, void* d_ws, size_t ws_size,
                              hipStream_t stream) {
  const float* x   = (const float*)d_in[0];
  const float* wm  = (const float*)d_in[1];
  const float* bm  = (const float*)d_in[2];
  const float* wa  = (const float*)d_in[3];
  const float* ba  = (const float*)d_in[4];
  const float* rot = (const float*)d_in[5];
  const float* wc  = (const float*)d_in[6];
  const float* bc  = (const float*)d_in[7];
  float* out = (float*)d_out;

  char* p = (char*)d_ws;
  size_t off = 0;
  auto alloc = [&](size_t bytes) { char* r = p + off; off = (off + bytes + 255) & ~(size_t)255; return r; };
  float*          xf0   = (float*)         alloc((size_t)L * C * 4);
  float*          fea1  = (float*)         alloc((size_t)L * C * 4);
  float*          fea2  = (float*)         alloc((size_t)L * C * 4);
  float*          q32   = (float*)         alloc((size_t)L * CR * 4);
  unsigned short* ybf   = (unsigned short*)alloc((size_t)L * C * 2);
  unsigned short* qAb   = (unsigned short*)alloc((size_t)NS * 32 * 2);
  unsigned short* qKb   = (unsigned short*)alloc((size_t)NS * 32 * 2);
  unsigned short* ykT   = (unsigned short*)alloc((size_t)NS * C * 2);
  float*          ret   = (float*)         alloc((size_t)NS * C * 4);
  float*          lseg  = (float*)         alloc((size_t)NS * 4);
  int*            codes = (int*)           alloc((size_t)NS * 4);
  int*            idx   = (int*)           alloc((size_t)NS * 4);
  int*            undo  = (int*)           alloc((size_t)NS * 4);
  int*            bh    = (int*)           alloc(128 * 128 * 4);
  int*            bst   = (int*)           alloc(128 * 128 * 4);

  const size_t attnLds = (1536 * 32 + 64 * 1536 + 8 * 512) * sizeof(unsigned short); // 303104

  k_transpose<<<L / 256, 256, 0, stream>>>(x, xf0);

  for (int lay = 0; lay < 2; ++lay) {
    const float* fin = lay ? fea1 : xf0;
    float* fout = lay ? fea2 : fea1;
    k_proj   <<<L / 256, 256, 0, stream>>>(fin, wm, bm, wa, ba, q32, ybf);
    k_hash   <<<L / 256, 256, 0, stream>>>(q32, rot + (size_t)lay * CR * NH * (HB / 2), codes);
    k_hist   <<<128, 512, 0, stream>>>(codes, bh);
    k_offsets<<<1, 128, 0, stream>>>(bh, bst);
    k_scatter<<<128, 512, 0, stream>>>(codes, bst, idx, undo);
    k_gather <<<NS / 256, 256, 0, stream>>>(idx, q32, ybf, qAb, qKb, ykT);
    k_attn   <<<128, 256, attnLds, stream>>>(qAb, qKb, ykT, ret, lseg);
    k_combine<<<L / 256, 256, 0, stream>>>(undo, ret, lseg, fin, fout);
  }

  k_collect<<<L / 128, 256, 0, stream>>>(fea1, fea2, wc, bc, out);
}
